// ImputationLoss_Missing_39075612459452
// MI455X (gfx1250) — matrix operand order repaired and checked by reading, not compiled, not run on hardware
//
#include <hip/hip_runtime.h>
#include <hip/hip_bf16.h>
#include <math.h>

// ---------------------------------------------------------------------------
// ImputationLoss (CE + minimac-r2 + masked pairwise JS "evo" loss) for MI455X
// B=32, L=16000, C=4, CF=5.  Output: single f32 scalar.
//
// Stage 1 (prep):   per (b,l): argmax label, CE nll, entropy, f16 indicator
//                   matrix a[b, l*4+c] = (lbl==c) for the msum GEMM.
// Stage 2 (r2):     group valid-count reduction, then per-locus r2 partials.
// Stage 3 (msum):   msum = a * a^T via v_wmma_f32_16x16x32_f16 (exact 0/1 GEMM).
// Stage 4 (pairs):  496 blocks, one per (i<j) pair: masked JS numerator.
// Stage 5 (final):  fixed-order partial sums + row-normalized evo weighting.
// All reductions are deterministic (block trees + fixed-order final sums;
// the only atomics add exact small integers in f32 -> order independent).
// ---------------------------------------------------------------------------

#define Bn  32
#define Ln  16000
#define Cn  4
#define CFn 5
#define Kn  (Ln * Cn)          // 64000
#define Gn  (Bn / 4)           // 8 groups

typedef __attribute__((ext_vector_type(16))) _Float16 v16h;
typedef __attribute__((ext_vector_type(8)))  _Float16 v8h;
typedef __attribute__((ext_vector_type(4)))  _Float16 v4h;
typedef __attribute__((ext_vector_type(8)))  float    v8f;

// Workspace byte offsets
#define OFF_ENT   ((size_t)0)                  // 512000 f32
#define OFF_LBL   ((size_t)2048000)            // 512000 i32
#define OFF_A16   ((size_t)4096000)            // 2048000 f16
#define OFF_MSUM  ((size_t)8192000)            // 1024 f32  (zeroed)
#define OFF_PAIR  ((size_t)(8192000 + 4096))   // 1024 f32  (zeroed)
#define OFF_CNT   ((size_t)(8192000 + 8192))   // 8 f32
#define OFF_PCE   (OFF_CNT + 32)               // 2000 f32
#define OFF_PR2   (OFF_PCE + 8000)             // 500 f32

__device__ __forceinline__ float block_reduce_256(float v, float* red) {
  int t = threadIdx.x;
  red[t] = v;
  __syncthreads();
  #pragma unroll
  for (int off = 128; off > 0; off >>= 1) {
    if (t < off) red[t] += red[t + off];
    __syncthreads();
  }
  float r = red[0];
  __syncthreads();
  return r;
}

// ---- Stage 1: per-(b,l) prep: label, CE, entropy, f16 indicator ------------
__global__ __launch_bounds__(256) void il_prep_kernel(
    const float* __restrict__ logits, const float* __restrict__ prob,
    const float* __restrict__ y_true, float* __restrict__ ent,
    int* __restrict__ lbl_out, _Float16* __restrict__ aF16,
    float* __restrict__ partial_ce) {
  __shared__ float red[256];
  int idx = blockIdx.x * 256 + threadIdx.x;   // [0, B*L) exactly

  // argmax over CF=5 one-hot columns (first max, matching jnp.argmax)
  const float* yt = y_true + (size_t)idx * CFn;
  int lbl = 0; float best = yt[0];
  #pragma unroll
  for (int c = 1; c < CFn; ++c) { float v = yt[c]; if (v > best) { best = v; lbl = c; } }
  lbl_out[idx] = lbl;
  bool valid = (lbl != Cn);

  // CE: -log_softmax(logits)[lbl], masked
  float4 x = ((const float4*)logits)[idx];
  float m   = fmaxf(fmaxf(x.x, x.y), fmaxf(x.z, x.w));
  float lse = logf(expf(x.x - m) + expf(x.y - m) + expf(x.z - m) + expf(x.w - m));
  float xl  = (lbl == 0) ? x.x : (lbl == 1) ? x.y : (lbl == 2) ? x.z : x.w;
  float nll = valid ? (m + lse - xl) : 0.0f;

  // entropy term: sum pc*log(pc), pc = max(prob, 1e-7)
  float4 p = ((const float4*)prob)[idx];
  float p0 = fmaxf(p.x, 1e-7f), p1 = fmaxf(p.y, 1e-7f);
  float p2 = fmaxf(p.z, 1e-7f), p3 = fmaxf(p.w, 1e-7f);
  ent[idx] = p0 * logf(p0) + p1 * logf(p1) + p2 * logf(p2) + p3 * logf(p3);

  // f16 indicator row chunk: a[b, l*4+c] = (lbl==c), c in 0..3
  v4h a;
  a[0] = (_Float16)(lbl == 0 ? 1.0f : 0.0f);
  a[1] = (_Float16)(lbl == 1 ? 1.0f : 0.0f);
  a[2] = (_Float16)(lbl == 2 ? 1.0f : 0.0f);
  a[3] = (_Float16)(lbl == 3 ? 1.0f : 0.0f);
  *(v4h*)(aF16 + (size_t)idx * 4) = a;

  float s = block_reduce_256(nll, red);
  if (threadIdx.x == 0) partial_ce[blockIdx.x] = s;
}

// ---- Stage 2a: per-group valid-locus count --------------------------------
__global__ __launch_bounds__(256) void il_r2cnt_kernel(
    const int* __restrict__ lbl, float* __restrict__ cnt) {
  __shared__ float red[256];
  int g = blockIdx.x;
  float s = 0.0f;
  for (int l = threadIdx.x; l < Ln; l += 256) {
    bool nm = false;
    #pragma unroll
    for (int mi = 0; mi < 4; ++mi) nm |= (lbl[(size_t)(g * 4 + mi) * Ln + l] != Cn);
    s += nm ? 1.0f : 0.0f;
  }
  float tot = block_reduce_256(s, red);
  if (threadIdx.x == 0) cnt[g] = fmaxf(4.0f * tot, 1.0f);
}

// ---- Stage 2b: per-(g,l) r2 partials --------------------------------------
__global__ __launch_bounds__(256) void il_r2_kernel(
    const int* __restrict__ lbl, const float* __restrict__ prob,
    const float* __restrict__ cnt, float* __restrict__ partial_r2) {
  __shared__ float red[256];
  int idx = blockIdx.x * 256 + threadIdx.x;   // [0, G*L) exactly (8*16000)
  int g = idx / Ln, l = idx - g * Ln;
  int alt = 0; bool nonmiss = false; float pred = 0.0f;
  #pragma unroll
  for (int mi = 0; mi < 4; ++mi) {
    int b = g * 4 + mi;
    int lb = lbl[(size_t)b * Ln + l];
    nonmiss |= (lb != Cn);
    alt += (lb >= 1 && lb <= 3) ? 1 : 0;
    float4 p = ((const float4*)prob)[(size_t)b * Ln + l];
    pred += p.y + p.z + p.w;
  }
  pred *= 0.25f;
  float af  = (float)alt / cnt[g];
  float af2 = nonmiss ? af : 0.5f;
  bool edge = (af2 == 0.0f) || (af2 == 1.0f);
  float denom = fmaxf(af2 * (1.0f - af2), 0.01f);
  float d = pred - af2;
  float r2 = (edge || !nonmiss) ? 0.0f : (d * d) / denom;
  float s = block_reduce_256(r2, red);
  if (threadIdx.x == 0) partial_r2[blockIdx.x] = s;
}

// ---- Stage 3: msum = a * a^T via WMMA f16 ---------------------------------
// 4 output tiles (16x16) x 16 K-chunks; exact 0/1 arithmetic -> f32 adds exact.
__global__ __launch_bounds__(32) void il_msum_wmma_kernel(
    const _Float16* __restrict__ aF16, float* __restrict__ msum) {
  const int NCHUNK = 16;
  const int KC = Kn / NCHUNK;                  // 4000, multiple of 32
  int tile  = blockIdx.x >> 4;                 // 0..3
  int chunk = blockIdx.x & 15;
  int ti = tile >> 1, tj = tile & 1;
  int t  = threadIdx.x;
  int hl = t >> 4;                             // half-wave: 0 or 1
  int ln = t & 15;
  const _Float16* Ar = aF16 + (size_t)(16 * ti + ln) * Kn;  // A row = M
  const _Float16* Br = aF16 + (size_t)(16 * tj + ln) * Kn;  // B col = N (a^T)
  v8f acc = {};
  int k0 = chunk * KC;
  for (int kb = k0; kb < k0 + KC; kb += 32) {
    // A fragment (16x32 f16): lane m holds K = kb+16h .. kb+16h+15 consecutive
    v8h alo = *(const v8h*)(Ar + kb + 16 * hl);
    v8h ahi = *(const v8h*)(Ar + kb + 16 * hl + 8);
    v16h a;
    #pragma unroll
    for (int e = 0; e < 8; ++e) { a[e] = alo[e]; a[e + 8] = ahi[e]; }
    // B fragment (32x16 f16): lane n holds K = kb+16h .. kb+16h+15 consecutive
    v16h b = *(const v16h*)(Br + kb + 16 * hl);
    acc = __builtin_amdgcn_wmma_f32_16x16x32_f16(
        false, a, false, b, (short)0, acc, false, false);
  }
  // C/D layout: VGPR v -> M = v + 8*half, N = lane%16
  #pragma unroll
  for (int v = 0; v < 8; ++v) {
    int m = 16 * ti + 8 * hl + v;
    int n = 16 * tj + ln;
    atomicAdd(&msum[m * Bn + n], acc[v]);      // exact integer adds
  }
}

// ---- Stage 4: masked pairwise JS numerators (one block per i<j pair) ------
__global__ __launch_bounds__(256) void il_pair_kernel(
    const float* __restrict__ prob, const float* __restrict__ ent,
    const int* __restrict__ lbl, float* __restrict__ pairfull) {
  __shared__ float red[256];
  int p = blockIdx.x;                // 0..495
  int i = 0, rem = p;
  while (rem >= 31 - i) { rem -= 31 - i; ++i; }
  int j = i + 1 + rem;

  const float4* pi4 = (const float4*)(prob + (size_t)i * Ln * Cn);
  const float4* pj4 = (const float4*)(prob + (size_t)j * Ln * Cn);
  const float* ei = ent + (size_t)i * Ln;
  const float* ej = ent + (size_t)j * Ln;
  const int*   li = lbl + (size_t)i * Ln;
  const int*   lj = lbl + (size_t)j * Ln;

  float num = 0.0f;
  for (int l = threadIdx.x; l < Ln; l += 256) {
    int a = li[l], b = lj[l];
    if (a == b && a != Cn) {                   // mask: both valid & equal
      float4 pa = pi4[l], pb = pj4[l];
      float s0 = fmaxf(pa.x, 1e-7f) + fmaxf(pb.x, 1e-7f);
      float s1 = fmaxf(pa.y, 1e-7f) + fmaxf(pb.y, 1e-7f);
      float s2 = fmaxf(pa.z, 1e-7f) + fmaxf(pb.z, 1e-7f);
      float s3 = fmaxf(pa.w, 1e-7f) + fmaxf(pb.w, 1e-7f);
      float cross = s0 * logf(0.5f * s0) + s1 * logf(0.5f * s1)
                  + s2 * logf(0.5f * s2) + s3 * logf(0.5f * s3);
      num += 0.5f * (ei[l] + ej[l] - cross);
    }
  }
  float s = block_reduce_256(num, red);
  if (threadIdx.x == 0) {
    pairfull[i * Bn + j] = s;                  // symmetric; diag stays 0
    pairfull[j * Bn + i] = s;
  }
}

// ---- Stage 5: fixed-order final combine -----------------------------------
__global__ __launch_bounds__(256) void il_final_kernel(
    const float* __restrict__ partial_ce, const float* __restrict__ partial_r2,
    const float* __restrict__ pairfull, const float* __restrict__ msum,
    const float* __restrict__ y_evo, float* __restrict__ out) {
  __shared__ float red[256];
  int t = threadIdx.x;

  float s = 0.0f;
  for (int k = t; k < 2000; k += 256) s += partial_ce[k];
  float ce = block_reduce_256(s, red);

  s = 0.0f;
  for (int k = t; k < 500; k += 256) s += partial_r2[k];
  float r2sum = block_reduce_256(s, red);

  float rowacc = 0.0f;
  if (t < Bn) {
    float wsum = 0.0f;
    for (int j = 0; j < Bn; ++j) wsum += expf(-1.0f * y_evo[t * Bn + j]);
    float inv = 1.0f / (wsum + 1e-8f);
    for (int j = 0; j < Bn; ++j) {
      float js = (t == j) ? 0.0f
                          : pairfull[t * Bn + j] / fmaxf(msum[t * Bn + j], 1.0f);
      rowacc += expf(-1.0f * y_evo[t * Bn + j]) * inv * js;
    }
  }
  float evo = block_reduce_256(rowacc, red);
  if (t == 0) {
    if (!isfinite(evo)) evo = 0.0f;
    float r2_loss = -(r2sum * 4.0f) / (float)Bn;   // R2_WEIGHT=1
    out[0] = ce + r2_loss + evo;                   // EVO_WEIGHT=1
  }
}

extern "C" void kernel_launch(void* const* d_in, const int* in_sizes, int n_in,
                              void* d_out, int out_size, void* d_ws, size_t ws_size,
                              hipStream_t stream) {
  const float* logits = (const float*)d_in[0];
  const float* prob   = (const float*)d_in[1];
  const float* y_true = (const float*)d_in[2];
  const float* y_evo  = (const float*)d_in[3];
  float* out = (float*)d_out;

  char* ws = (char*)d_ws;
  float*    ent      = (float*)(ws + OFF_ENT);
  int*      lbl      = (int*)(ws + OFF_LBL);
  _Float16* aF16     = (_Float16*)(ws + OFF_A16);
  float*    msum     = (float*)(ws + OFF_MSUM);
  float*    pairfull = (float*)(ws + OFF_PAIR);
  float*    cnt      = (float*)(ws + OFF_CNT);
  float*    pce      = (float*)(ws + OFF_PCE);
  float*    pr2      = (float*)(ws + OFF_PR2);

  // zero msum + pairfull (contiguous 8192 bytes)
  hipMemsetAsync(ws + OFF_MSUM, 0, 8192, stream);

  il_prep_kernel<<<(Bn * Ln) / 256, 256, 0, stream>>>(
      logits, prob, y_true, ent, lbl, aF16, pce);
  il_r2cnt_kernel<<<Gn, 256, 0, stream>>>(lbl, cnt);
  il_r2_kernel<<<(Gn * Ln) / 256, 256, 0, stream>>>(lbl, prob, cnt, pr2);
  il_msum_wmma_kernel<<<64, 32, 0, stream>>>(aF16, msum);
  il_pair_kernel<<<496, 256, 0, stream>>>(prob, ent, lbl, pairfull);
  il_final_kernel<<<1, 256, 0, stream>>>(pce, pr2, pairfull, msum, y_evo, out);
}
